// LFADS_77395310674198
// MI455X (gfx1250) — matrix operand order repaired and checked by reading, not compiled, not run on hardware
//
#include <hip/hip_runtime.h>
#include <math.h>

// ---------------------------------------------------------------------------
// LFADS forward for MI455X (gfx1250, wave32, WMMA f32<-f16 16x16x32)
// ---------------------------------------------------------------------------

typedef __attribute__((ext_vector_type(16))) _Float16 v16h;
typedef __attribute__((ext_vector_type(8)))  float    v8f;

#define WMMA_F16(a, b, c) \
  __builtin_amdgcn_wmma_f32_16x16x32_f16(false, (a), false, (b), (short)0, (c), false, false)

// Model dims
#define BB   64
#define TT   512
#define NN   256
#define HH   256      // hidden for all GRUs (C_DIM == G_DIM == ENC_* == 256)
#define G3   768      // 3*H gate width
#define UD   128
#define FD   128
#define MT   (BB * TT)

__device__ __forceinline__ float sigm(float x) { return 1.0f / (1.0f + __expf(-x)); }
__device__ __forceinline__ float clip5(float x) { return fminf(fmaxf(x, -5.0f), 5.0f); }

// ---- WMMA fragment loaders (wave32 layouts per CDNA5 ISA 7.12.2) -----------
// A: logical 16x32 (MxK). lane: m = lane&15 ; element i -> k = (i&8)*2 + (lane>>4)*8 + (i&7)
__device__ __forceinline__ v16h load_a_f16(const _Float16* A, int lda, int row0, int k0, int lane) {
  const _Float16* p = A + (size_t)(row0 + (lane & 15)) * lda + k0 + ((lane >> 4) << 3);
  v16h a;
#pragma unroll
  for (int i = 0; i < 8; ++i) a[i] = p[i];
#pragma unroll
  for (int i = 0; i < 8; ++i) a[8 + i] = p[16 + i];
  return a;
}
__device__ __forceinline__ v16h load_a_f32(const float* A, int lda, int row0, int k0, int lane) {
  const float* p = A + (size_t)(row0 + (lane & 15)) * lda + k0 + ((lane >> 4) << 3);
  v16h a;
#pragma unroll
  for (int i = 0; i < 8; ++i) a[i] = (_Float16)p[i];
#pragma unroll
  for (int i = 0; i < 8; ++i) a[8 + i] = (_Float16)p[16 + i];
  return a;
}
// B: logical 32x16 (KxN) = W^T tile, W row-major (N x K, row stride ldw).
// lane: n = lane&15 ; element i -> k = k0 + (i&8)*2 + (lane>>4)*8 + (i&7)  (same as A)
__device__ __forceinline__ v16h load_b_f16(const _Float16* W, int ldw, int n0, int k0, int lane) {
  const _Float16* p = W + (size_t)(n0 + (lane & 15)) * ldw + k0 + ((lane >> 4) << 3);
  v16h b;
#pragma unroll
  for (int i = 0; i < 8; ++i) b[i] = p[i];
#pragma unroll
  for (int i = 0; i < 8; ++i) b[8 + i] = p[16 + i];
  return b;
}

// ---------------------------------------------------------------------------
// Generic tiled GEMM:  C(M,N) = act( A(M,K) @ W(N,K)^T + bias )
// One wave per 16x16 output tile. OMODE: 0=f32, 1=f32+exp, 2=f16, 3=f16+clip5
// ---------------------------------------------------------------------------
template <bool AF16, int OMODE>
__global__ void proj_gemm_kernel(const void* A_, const _Float16* W, const float* bias,
                                 void* C_, int M, int N, int K, int lda, int ldw, int ldc) {
  const int lane = threadIdx.x & 31;
  const int wid  = blockIdx.x * (blockDim.x >> 5) + (threadIdx.x >> 5);
  const int tiles_n = N >> 4;
  const int tm = wid / tiles_n, tn = wid % tiles_n;
  if (tm >= (M >> 4)) return;

  v8f acc = {};
  for (int k0 = 0; k0 < K; k0 += 32) {
    v16h a;
    if (AF16) a = load_a_f16((const _Float16*)A_, lda, tm * 16, k0, lane);
    else      a = load_a_f32((const float*)A_,    lda, tm * 16, k0, lane);
    v16h b = load_b_f16(W, ldw, tn * 16, k0, lane);
    acc = WMMA_F16(a, b, acc);
  }
  const int n = tn * 16 + (lane & 15);
  const float bv = bias ? bias[n] : 0.0f;
#pragma unroll
  for (int r = 0; r < 8; ++r) {
    const int m = tm * 16 + r + ((lane >> 4) << 3);
    float v = acc[r] + bv;
    if (OMODE == 1) v = __expf(v);
    if (OMODE == 3) v = clip5(v);
    if (OMODE <= 1) ((float*)C_)[(size_t)m * ldc + n] = v;
    else            ((_Float16*)C_)[(size_t)m * ldc + n] = (_Float16)v;
  }
}

__global__ void cvt_f16_kernel(const float* src, _Float16* dst, int n) {
  int i = blockIdx.x * blockDim.x + threadIdx.x;
  if (i < n) dst[i] = (_Float16)src[i];
}

__global__ void g0_kernel(const float* stats, const float* eps, float* g0) {
  int i = blockIdx.x * blockDim.x + threadIdx.x;
  if (i >= BB * HH) return;
  int b = i >> 8, n = i & 255;
  float mean = stats[b * 512 + n];
  float logv = stats[b * 512 + 256 + n];
  g0[i] = eps[i] * __expf(0.5f * logv) + mean;
}

// ---------------------------------------------------------------------------
// Encoder GRU scans: 4 independent persistent workgroups (g_fwd, g_bwd, c_fwd,
// c_bwd). Hidden state double-buffered in LDS as f16 (also the WMMA A operand).
// Per step: hg = h @ W_hh^T (64x768 <- 64x256 @ 256x768) fused with gates.
// ---------------------------------------------------------------------------
struct EncArgs {
  const _Float16* xg[4];   // precomputed x@W_ih^T + b_ih, f16, [B*T, 768]
  const _Float16* whh[4];  // f16, [768, 256]
  const float*    bhh[4];
  float*          henc;    // [64, 512] clipped final h for jobs 0,1
  _Float16*       outc;    // [B*T, 512] clipped h_t for jobs 2,3
};

__global__ void enc_scan_kernel(EncArgs args) {
  extern __shared__ _Float16 smem_es[];
  _Float16* h0 = smem_es;            // [64*256]
  _Float16* h1 = smem_es + 64 * 256; // [64*256]

  const int job  = blockIdx.x;
  const bool rev = (job & 1);
  const _Float16* xg  = args.xg[job];
  const _Float16* whh = args.whh[job];
  const float*    bhh = args.bhh[job];
  const int tid = threadIdx.x, lane = tid & 31, w = tid >> 5;

  for (int i = tid; i < 64 * 256; i += blockDim.x) { h0[i] = (_Float16)0.f; h1[i] = (_Float16)0.f; }
  __syncthreads();

  for (int s = 0; s < TT; ++s) {
    const int t = rev ? (TT - 1 - s) : s;
    _Float16* hc = (s & 1) ? h1 : h0;
    _Float16* hn = (s & 1) ? h0 : h1;
    // 64 tile-triples (4 m-tiles x 16 n-tiles), 32 waves -> 2 each
    for (int q = w * 2; q < w * 2 + 2; ++q) {
      const int tm = q >> 4, tn = q & 15;
      v8f ar = {}, az = {}, an = {};
      for (int k0 = 0; k0 < HH; k0 += 32) {
        v16h a  = load_a_f16(hc, HH, tm * 16, k0, lane);
        v16h br = load_b_f16(whh, HH, 0 * HH + tn * 16, k0, lane);
        v16h bz = load_b_f16(whh, HH, 1 * HH + tn * 16, k0, lane);
        v16h bn = load_b_f16(whh, HH, 2 * HH + tn * 16, k0, lane);
        ar = WMMA_F16(a, br, ar);
        az = WMMA_F16(a, bz, az);
        an = WMMA_F16(a, bn, an);
      }
      const int n = tn * 16 + (lane & 15);
      const float bhr = bhh[n], bhz = bhh[HH + n], bhn = bhh[2 * HH + n];
#pragma unroll
      for (int r = 0; r < 8; ++r) {
        const int m = tm * 16 + r + ((lane >> 4) << 3);
        const size_t row = (size_t)m * TT + t;
        const float xr = (float)xg[row * G3 + n];
        const float xz = (float)xg[row * G3 + HH + n];
        const float xn = (float)xg[row * G3 + 2 * HH + n];
        const float rg = sigm(xr + ar[r] + bhr);
        const float zg = sigm(xz + az[r] + bhz);
        const float nn = tanhf(xn + rg * (an[r] + bhn));
        const float hold = (float)hc[m * HH + n];
        const float hv = (1.0f - zg) * nn + zg * hold;
        hn[m * HH + n] = (_Float16)hv;
        if (job >= 2) args.outc[row * 512 + (job - 2) * HH + n] = (_Float16)clip5(hv);
      }
    }
    __syncthreads();
  }
  if (job < 2) { // final h (in h0: 512 steps, even) -> clipped into henc concat
    for (int i = tid; i < 64 * 256; i += blockDim.x) {
      const int m = i >> 8, n = i & 255;
      args.henc[(size_t)m * 512 + job * HH + n] = clip5((float)h0[i]);
    }
  }
}

// ---------------------------------------------------------------------------
// Generator/controller scan: one persistent workgroup, 4 WMMA phases/step with
// barriers. h_c and g double-buffered f16 in LDS; u, f single-buffered.
// (No arrays of LDS pointers: computed offsets only, to avoid addrspacecast
//  static initializers that ld.lld rejects.)
// ---------------------------------------------------------------------------
struct GenArgs {
  const _Float16* encpre;   // f16 [B*T, 768] = out_enc_c@W_enc^T + bctl_ih
  const float*    epsu;     // f32 [B, T, 128]
  const float*    g0;       // f32 [64, 256]
  const _Float16* wctlih;   // f16 [768, 640]  (W_f = cols 512..639)
  const _Float16* wctlhh;   // f16 [768, 256]
  const float*    bctlhh;
  const _Float16* wu;       // f16 [256, 256]
  const float*    bu;
  const _Float16* wgenih;   // f16 [768, 128]
  const float*    bgenih;
  const _Float16* wgenhh;   // f16 [768, 256]
  const float*    bgenhh;
  const _Float16* wfac;     // f16 [128, 256]
  const float*    bfac;
  _Float16*       factors;  // f16 [B*T, 128]
};

__global__ void gen_scan_kernel(GenArgs args) {
  extern __shared__ _Float16 smem_gs[];
  // layout: [hc0 | hc1 | g0buf | g1buf | u | f]
  _Float16* ub = smem_gs + 4 * 64 * 256;  // [64*128]
  _Float16* fb = ub + 64 * 128;           // [64*128]
  const int tid = threadIdx.x, lane = tid & 31, w = tid >> 5;

  for (int i = tid; i < 64 * 256; i += blockDim.x) {
    smem_gs[0 * 64 * 256 + i] = (_Float16)0.f;         // hc buf 0
    smem_gs[1 * 64 * 256 + i] = (_Float16)0.f;         // hc buf 1
    smem_gs[2 * 64 * 256 + i] = (_Float16)args.g0[i];  // g buf 0
    smem_gs[3 * 64 * 256 + i] = (_Float16)0.f;         // g buf 1
  }
  __syncthreads();
  int hcur = 0, gcur = 0;

  // f0 = g0 @ W_fac^T + b_fac (LDS only)
  if (w < 32) {
    const _Float16* gsrc = smem_gs + (2 + gcur) * 64 * 256;
    const int tm = w >> 3, tn = w & 7;
    v8f acc = {};
    for (int k0 = 0; k0 < HH; k0 += 32) {
      v16h a = load_a_f16(gsrc, HH, tm * 16, k0, lane);
      v16h b = load_b_f16(args.wfac, HH, tn * 16, k0, lane);
      acc = WMMA_F16(a, b, acc);
    }
    const int n = tn * 16 + (lane & 15);
    const float bf = args.bfac[n];
#pragma unroll
    for (int r = 0; r < 8; ++r) {
      const int m = tm * 16 + r + ((lane >> 4) << 3);
      fb[m * FD + n] = (_Float16)(acc[r] + bf);
    }
  }
  __syncthreads();

  for (int t = 0; t < TT; ++t) {
    // ---- P1: controller GRU: gates from f@W_f^T + enc_pre (x side) and h_c@Wctl_hh^T
    {
      const _Float16* hsrc = smem_gs + hcur * 64 * 256;
      _Float16*       hdst = smem_gs + (hcur ^ 1) * 64 * 256;
      for (int q = w * 2; q < w * 2 + 2; ++q) {
        const int tm = q >> 4, tn = q & 15;
        v8f ar = {}, az = {}, axn = {}, ahn = {};
        for (int k0 = 0; k0 < FD; k0 += 32) {          // x-side: f @ W_f^T
          v16h a  = load_a_f16(fb, FD, tm * 16, k0, lane);
          v16h br = load_b_f16(args.wctlih, 640, 0 * HH + tn * 16, 512 + k0, lane);
          v16h bz = load_b_f16(args.wctlih, 640, 1 * HH + tn * 16, 512 + k0, lane);
          v16h bn = load_b_f16(args.wctlih, 640, 2 * HH + tn * 16, 512 + k0, lane);
          ar = WMMA_F16(a, br, ar); az = WMMA_F16(a, bz, az); axn = WMMA_F16(a, bn, axn);
        }
        for (int k0 = 0; k0 < HH; k0 += 32) {          // h-side: h_c @ Wctl_hh^T
          v16h a  = load_a_f16(hsrc, HH, tm * 16, k0, lane);
          v16h br = load_b_f16(args.wctlhh, HH, 0 * HH + tn * 16, k0, lane);
          v16h bz = load_b_f16(args.wctlhh, HH, 1 * HH + tn * 16, k0, lane);
          v16h bn = load_b_f16(args.wctlhh, HH, 2 * HH + tn * 16, k0, lane);
          ar = WMMA_F16(a, br, ar); az = WMMA_F16(a, bz, az); ahn = WMMA_F16(a, bn, ahn);
        }
        const int n = tn * 16 + (lane & 15);
        const float bhr = args.bctlhh[n], bhz = args.bctlhh[HH + n], bhn = args.bctlhh[2 * HH + n];
#pragma unroll
        for (int r = 0; r < 8; ++r) {
          const int m = tm * 16 + r + ((lane >> 4) << 3);
          const size_t row = (size_t)m * TT + t;
          const float er = (float)args.encpre[row * G3 + n];
          const float ez = (float)args.encpre[row * G3 + HH + n];
          const float en = (float)args.encpre[row * G3 + 2 * HH + n];
          const float rg = sigm(er + ar[r] + bhr);
          const float zg = sigm(ez + az[r] + bhz);
          const float nn = tanhf(en + axn[r] + rg * (ahn[r] + bhn));
          const float hold = (float)hsrc[m * HH + n];
          hdst[m * HH + n] = (_Float16)clip5((1.0f - zg) * nn + zg * hold);
        }
      }
    }
    __syncthreads();
    hcur ^= 1;
    // ---- P2: u = eps * exp(0.5*logv) + mean,  [mean|logv] = h_c @ W_u^T + b_u
    if (w < 32) {
      const _Float16* hsrc = smem_gs + hcur * 64 * 256;
      const int tm = w >> 3, tn = w & 7;
      v8f am = {}, av = {};
      for (int k0 = 0; k0 < HH; k0 += 32) {
        v16h a  = load_a_f16(hsrc, HH, tm * 16, k0, lane);
        v16h bm = load_b_f16(args.wu, HH, tn * 16, k0, lane);
        v16h bv = load_b_f16(args.wu, HH, UD + tn * 16, k0, lane);
        am = WMMA_F16(a, bm, am); av = WMMA_F16(a, bv, av);
      }
      const int n = tn * 16 + (lane & 15);
      const float bm_ = args.bu[n], bv_ = args.bu[UD + n];
#pragma unroll
      for (int r = 0; r < 8; ++r) {
        const int m = tm * 16 + r + ((lane >> 4) << 3);
        const float mean = am[r] + bm_, logv = av[r] + bv_;
        const float e = args.epsu[((size_t)m * TT + t) * UD + n];
        ub[m * UD + n] = (_Float16)(e * __expf(0.5f * logv) + mean);
      }
    }
    __syncthreads();
    // ---- P3: generator GRU: u@Wgen_ih^T (x side) and g@Wgen_hh^T (h side)
    {
      const _Float16* gsrc = smem_gs + (2 + gcur) * 64 * 256;
      _Float16*       gdst = smem_gs + (2 + (gcur ^ 1)) * 64 * 256;
      for (int q = w * 2; q < w * 2 + 2; ++q) {
        const int tm = q >> 4, tn = q & 15;
        v8f ar = {}, az = {}, axn = {}, ahn = {};
        for (int k0 = 0; k0 < UD; k0 += 32) {
          v16h a  = load_a_f16(ub, UD, tm * 16, k0, lane);
          v16h br = load_b_f16(args.wgenih, UD, 0 * HH + tn * 16, k0, lane);
          v16h bz = load_b_f16(args.wgenih, UD, 1 * HH + tn * 16, k0, lane);
          v16h bn = load_b_f16(args.wgenih, UD, 2 * HH + tn * 16, k0, lane);
          ar = WMMA_F16(a, br, ar); az = WMMA_F16(a, bz, az); axn = WMMA_F16(a, bn, axn);
        }
        for (int k0 = 0; k0 < HH; k0 += 32) {
          v16h a  = load_a_f16(gsrc, HH, tm * 16, k0, lane);
          v16h br = load_b_f16(args.wgenhh, HH, 0 * HH + tn * 16, k0, lane);
          v16h bz = load_b_f16(args.wgenhh, HH, 1 * HH + tn * 16, k0, lane);
          v16h bn = load_b_f16(args.wgenhh, HH, 2 * HH + tn * 16, k0, lane);
          ar = WMMA_F16(a, br, ar); az = WMMA_F16(a, bz, az); ahn = WMMA_F16(a, bn, ahn);
        }
        const int n = tn * 16 + (lane & 15);
        const float bir = args.bgenih[n], biz = args.bgenih[HH + n], bin_ = args.bgenih[2 * HH + n];
        const float bhr = args.bgenhh[n], bhz = args.bgenhh[HH + n], bhn = args.bgenhh[2 * HH + n];
#pragma unroll
        for (int r = 0; r < 8; ++r) {
          const int m = tm * 16 + r + ((lane >> 4) << 3);
          const float rg = sigm(ar[r] + bir + bhr);
          const float zg = sigm(az[r] + biz + bhz);
          const float nn = tanhf(axn[r] + bin_ + rg * (ahn[r] + bhn));
          const float gold = (float)gsrc[m * HH + n];
          gdst[m * HH + n] = (_Float16)clip5((1.0f - zg) * nn + zg * gold);
        }
      }
    }
    __syncthreads();
    gcur ^= 1;
    // ---- P4: f = g @ W_fac^T + b_fac  -> LDS (next step) + factors[t]
    if (w < 32) {
      const _Float16* gsrc = smem_gs + (2 + gcur) * 64 * 256;
      const int tm = w >> 3, tn = w & 7;
      v8f acc = {};
      for (int k0 = 0; k0 < HH; k0 += 32) {
        v16h a = load_a_f16(gsrc, HH, tm * 16, k0, lane);
        v16h b = load_b_f16(args.wfac, HH, tn * 16, k0, lane);
        acc = WMMA_F16(a, b, acc);
      }
      const int n = tn * 16 + (lane & 15);
      const float bf = args.bfac[n];
#pragma unroll
      for (int r = 0; r < 8; ++r) {
        const int m = tm * 16 + r + ((lane >> 4) << 3);
        const float fv = acc[r] + bf;
        fb[m * FD + n] = (_Float16)fv;
        args.factors[((size_t)m * TT + t) * FD + n] = (_Float16)fv;
      }
    }
    __syncthreads();
  }
}

// ---------------------------------------------------------------------------
// Host orchestration
// ---------------------------------------------------------------------------
extern "C" void kernel_launch(void* const* d_in, const int* in_sizes, int n_in,
                              void* d_out, int out_size, void* d_ws, size_t ws_size,
                              hipStream_t stream) {
  (void)in_sizes; (void)n_in; (void)out_size; (void)ws_size;

  const float* x = (const float*)d_in[0];

  // ---- workspace carve (256B aligned) ----
  char* p = (char*)d_ws;
  auto take = [&](size_t bytes) -> void* {
    void* r = (void*)p;
    p += (bytes + 255) & ~(size_t)255;
    return r;
  };
  _Float16* wih16[4], *whh16[4];
  for (int e = 0; e < 4; ++e) wih16[e] = (_Float16*)take((size_t)G3 * NN * 2);
  for (int e = 0; e < 4; ++e) whh16[e] = (_Float16*)take((size_t)G3 * HH * 2);
  _Float16* wg0_16    = (_Float16*)take((size_t)512 * 512 * 2);
  _Float16* wctlih16  = (_Float16*)take((size_t)G3 * 640 * 2);
  _Float16* wctlhh16  = (_Float16*)take((size_t)G3 * HH * 2);
  _Float16* wu16      = (_Float16*)take((size_t)256 * 256 * 2);
  _Float16* wgenih16  = (_Float16*)take((size_t)G3 * UD * 2);
  _Float16* wgenhh16  = (_Float16*)take((size_t)G3 * HH * 2);
  _Float16* wfac16    = (_Float16*)take((size_t)FD * HH * 2);
  _Float16* wrate16   = (_Float16*)take((size_t)NN * FD * 2);
  _Float16* xg16[4];
  for (int e = 0; e < 4; ++e) xg16[e] = (_Float16*)take((size_t)MT * G3 * 2);
  _Float16* outc16    = (_Float16*)take((size_t)MT * 512 * 2);
  _Float16* encpre16  = (_Float16*)take((size_t)MT * G3 * 2);
  _Float16* factors16 = (_Float16*)take((size_t)MT * FD * 2);
  float* henc    = (float*)take((size_t)BB * 512 * 4);
  float* g0stats = (float*)take((size_t)BB * 512 * 4);
  float* g0      = (float*)take((size_t)BB * HH * 4);

  // ---- 1. convert weights to f16 ----
  auto cvt = [&](int idx, _Float16* dst, int n) {
    cvt_f16_kernel<<<(n + 255) / 256, 256, 0, stream>>>((const float*)d_in[idx], dst, n);
  };
  const int wih_src[4] = {3, 7, 11, 15}, whh_src[4] = {4, 8, 12, 16};
  for (int e = 0; e < 4; ++e) { cvt(wih_src[e], wih16[e], G3 * NN); cvt(whh_src[e], whh16[e], G3 * HH); }
  cvt(19, wg0_16, 512 * 512);
  cvt(21, wctlih16, G3 * 640);
  cvt(22, wctlhh16, G3 * HH);
  cvt(25, wu16, 256 * 256);
  cvt(27, wgenih16, G3 * UD);
  cvt(28, wgenhh16, G3 * HH);
  cvt(31, wfac16, FD * HH);
  cvt(33, wrate16, NN * FD);

  // ---- 2. input projections xg_e = x @ W_ih_e^T + b_ih_e  (f16 out) ----
  const int bih_src[4] = {5, 9, 13, 17};
  const int xg_blocks = (MT / 16) * (G3 / 16) / 8;  // 8 waves / block
  for (int e = 0; e < 4; ++e) {
    proj_gemm_kernel<false, 2><<<xg_blocks, 256, 0, stream>>>(
        x, wih16[e], (const float*)d_in[bih_src[e]], xg16[e],
        MT, G3, NN, NN, NN, G3);
  }

  // ---- 3. encoder scans (4 persistent workgroups) ----
  EncArgs ea;
  const int bhh_src[4] = {6, 10, 14, 18};
  for (int e = 0; e < 4; ++e) {
    ea.xg[e] = xg16[e]; ea.whh[e] = whh16[e]; ea.bhh[e] = (const float*)d_in[bhh_src[e]];
  }
  ea.henc = henc; ea.outc = outc16;
  enc_scan_kernel<<<4, 1024, 2 * 64 * 256 * sizeof(_Float16), stream>>>(ea);

  // ---- 4. g0 stats + reparameterization ----
  proj_gemm_kernel<false, 0><<<(BB / 16) * (512 / 16) / 8, 256, 0, stream>>>(
      henc, wg0_16, (const float*)d_in[20], g0stats, BB, 512, 512, 512, 512, 512);
  g0_kernel<<<(BB * HH + 255) / 256, 256, 0, stream>>>(g0stats, (const float*)d_in[1], g0);

  // ---- 5. enc_pre = out_enc_c @ W_enc^T + bctl_ih  (W_enc = Wctl_ih[:, :512]) ----
  proj_gemm_kernel<true, 2><<<(MT / 16) * (G3 / 16) / 8, 256, 0, stream>>>(
      outc16, wctlih16, (const float*)d_in[23], encpre16, MT, G3, 512, 512, 640, G3);

  // ---- 6. generator/controller scan (one persistent workgroup) ----
  GenArgs ga;
  ga.encpre = encpre16; ga.epsu = (const float*)d_in[2]; ga.g0 = g0;
  ga.wctlih = wctlih16; ga.wctlhh = wctlhh16; ga.bctlhh = (const float*)d_in[24];
  ga.wu = wu16; ga.bu = (const float*)d_in[26];
  ga.wgenih = wgenih16; ga.bgenih = (const float*)d_in[29];
  ga.wgenhh = wgenhh16; ga.bgenhh = (const float*)d_in[30];
  ga.wfac = wfac16; ga.bfac = (const float*)d_in[32];
  ga.factors = factors16;
  const size_t gen_lds = (4 * 64 * 256 + 2 * 64 * 128) * sizeof(_Float16);  // 160 KB
  gen_scan_kernel<<<1, 1024, gen_lds, stream>>>(ga);

  // ---- 7. rates = exp(factors @ W_rate^T + b_rate) -> d_out ----
  proj_gemm_kernel<true, 1><<<(MT / 16) * (NN / 16) / 8, 256, 0, stream>>>(
      factors16, wrate16, (const float*)d_in[34], d_out, MT, NN, FD, FD, FD, NN);
}
